// MTGNN_18863496364080
// MI455X (gfx1250) — matrix operand order repaired and checked by reading, not compiled, not run on hardware
//
#include <hip/hip_runtime.h>
#include <hip/hip_bf16.h>

// ---------------------------------------------------------------------------
// MTGNN layer forward for MI455X (gfx1250, wave32, WMMA + TDM).
// Heavy op = mixprop graph diffusion -> v_wmma_f32_16x16x32_bf16,
// A-operand slab staged to LDS by the Tensor Data Mover.
// ---------------------------------------------------------------------------

#define Bb   32
#define Cc   32
#define Nn   300
#define Tt   64
#define TOUTc 58
#define DIMe 40
#define KTOP 20
#define VP   320   // padded node count (rows and K dim), 320 = 10*32 = 20*16
#define TP   64    // padded time

typedef __attribute__((ext_vector_type(16))) __bf16       v16bf;
typedef __attribute__((ext_vector_type(8)))  float        v8f;
typedef __attribute__((ext_vector_type(8)))  unsigned int v8u;
typedef __attribute__((ext_vector_type(4)))  unsigned int v4u;
typedef __attribute__((ext_vector_type(8)))  int          v8i;
typedef __attribute__((ext_vector_type(4)))  int          v4i;

#if __has_builtin(__builtin_amdgcn_tensor_load_to_lds)
#define HAVE_TDM 1
#endif

__device__ __forceinline__ float bf2f(unsigned short u) {
    unsigned int x = ((unsigned int)u) << 16;
    return __builtin_bit_cast(float, x);
}
__device__ __forceinline__ unsigned short f2bf(float f) {
    unsigned int x = __builtin_bit_cast(unsigned int, f);
    unsigned int r = x + 0x7fffu + ((x >> 16) & 1u);   // round-to-nearest-even
    return (unsigned short)(r >> 16);
}

// ---------------------------------------------------------------------------
// 1) Graph constructor: n1/n2 = tanh(3*(emb @ W^T + b))
// ---------------------------------------------------------------------------
__global__ __launch_bounds__(64)
void embed_kernel(const float* __restrict__ emb1, const float* __restrict__ emb2,
                  const float* __restrict__ l1w, const float* __restrict__ l1b,
                  const float* __restrict__ l2w, const float* __restrict__ l2b,
                  const int* __restrict__ idx,
                  float* __restrict__ n1, float* __restrict__ n2) {
    int i = blockIdx.x;
    int d = threadIdx.x;
    if (d >= DIMe) return;
    int ii = idx[i];
    float a = l1b[d], b = l2b[d];
    for (int e = 0; e < DIMe; ++e) {
        a += emb1[ii * DIMe + e] * l1w[d * DIMe + e];
        b += emb2[ii * DIMe + e] * l2w[d * DIMe + e];
    }
    n1[i * DIMe + d] = tanhf(3.0f * a);
    n2[i * DIMe + d] = tanhf(3.0f * b);
}

// A = relu(tanh(3*(n1 n2^T - n2 n1^T)))
__global__ __launch_bounds__(128)
void adj_kernel(const float* __restrict__ n1, const float* __restrict__ n2,
                float* __restrict__ A) {
    int i = blockIdx.x;
    __shared__ float r1[DIMe], r2[DIMe];
    if (threadIdx.x < DIMe) {
        r1[threadIdx.x] = n1[i * DIMe + threadIdx.x];
        r2[threadIdx.x] = n2[i * DIMe + threadIdx.x];
    }
    __syncthreads();
    for (int j = threadIdx.x; j < Nn; j += 128) {
        float a = 0.f;
        for (int e = 0; e < DIMe; ++e)
            a += r1[e] * n2[j * DIMe + e] - r2[e] * n1[j * DIMe + e];
        float t = tanhf(3.0f * a);
        A[i * Nn + j] = t > 0.f ? t : 0.f;
    }
}

// Per-row top-20 mask (iterative argmax, lowest-index tiebreak like lax.top_k)
__global__ __launch_bounds__(128)
void topk_kernel(const float* __restrict__ A, float* __restrict__ Am) {
    int i = blockIdx.x, tid = threadIdx.x;
    __shared__ float val[Nn];
    __shared__ int   sel[Nn];
    __shared__ float bv[128];
    __shared__ int   bi[128];
    for (int j = tid; j < Nn; j += 128) { val[j] = A[i * Nn + j]; sel[j] = 0; }
    __syncthreads();
    for (int k = 0; k < KTOP; ++k) {
        float mv = -1e30f; int mi = 0x7fffffff;
        for (int j = tid; j < Nn; j += 128)
            if (!sel[j] && (val[j] > mv || (val[j] == mv && j < mi))) { mv = val[j]; mi = j; }
        bv[tid] = mv; bi[tid] = mi;
        __syncthreads();
        for (int st = 64; st > 0; st >>= 1) {
            if (tid < st) {
                if (bv[tid + st] > bv[tid] ||
                    (bv[tid + st] == bv[tid] && bi[tid + st] < bi[tid])) {
                    bv[tid] = bv[tid + st]; bi[tid] = bi[tid + st];
                }
            }
            __syncthreads();
        }
        if (tid == 0 && bi[0] < Nn) sel[bi[0]] = 1;
        __syncthreads();
    }
    for (int j = tid; j < Nn; j += 128) Am[i * Nn + j] = sel[j] ? val[j] : 0.f;
}

__global__ __launch_bounds__(128)
void rowsum_kernel(const float* __restrict__ Am,
                   float* __restrict__ rs1, float* __restrict__ rs2) {
    int v = blockIdx.x, tid = threadIdx.x;
    float s1 = 0.f, s2 = 0.f;
    for (int j = tid; j < Nn; j += 128) { s1 += Am[v * Nn + j]; s2 += Am[j * Nn + v]; }
    __shared__ float a[128], b[128];
    a[tid] = s1; b[tid] = s2;
    __syncthreads();
    for (int st = 64; st > 0; st >>= 1) {
        if (tid < st) { a[tid] += a[tid + st]; b[tid] += b[tid + st]; }
        __syncthreads();
    }
    if (tid == 0) { rs1[v] = a[0] + 1.f; rs2[v] = b[0] + 1.f; }
}

// Ah1 = rownorm(A + I), Ah2 = rownorm(A^T + I), bf16, zero-padded to VPxVP
__global__ __launch_bounds__(64)
void writeAh_kernel(const float* __restrict__ Am,
                    const float* __restrict__ rs1, const float* __restrict__ rs2,
                    unsigned short* __restrict__ Ah1, unsigned short* __restrict__ Ah2) {
    int w = blockIdx.x * 64 + threadIdx.x;
    int v = blockIdx.y;
    if (w >= VP) return;
    float a1 = 0.f, a2 = 0.f;
    if (v < Nn && w < Nn) {
        float d = (v == w) ? 1.f : 0.f;
        a1 = (Am[v * Nn + w] + d) / rs1[v];
        a2 = (Am[w * Nn + v] + d) / rs2[v];
    }
    Ah1[v * VP + w] = f2bf(a1);
    Ah2[v * VP + w] = f2bf(a2);
}

// ---------------------------------------------------------------------------
// 2) Dilated inception + gating -> X bf16 in padded [b][c][VP][TP]
// ---------------------------------------------------------------------------
struct IncParams {
    const float* fw[4]; const float* fb[4];
    const float* gw[4]; const float* gb[4];
};

__global__ __launch_bounds__(256)
void inception_kernel(const float* __restrict__ x, IncParams p,
                      unsigned short* __restrict__ Xbf) {
    const int ksz[4] = {2, 3, 6, 7};
    int b = blockIdx.x;
    int n = blockIdx.y;                       // 0..VP-1 (pad rows write zeros)
    __shared__ float xs[Cc * Tt];             // 8 KB
    int tid = threadIdx.x;
    if (n < Nn) {
        for (int i = tid; i < Cc * Tt; i += 256) {
            int ic = i >> 6, t = i & 63;
            xs[i] = x[(((size_t)b * Cc + ic) * Nn + n) * Tt + t];
        }
    }
    __syncthreads();
    int oc = tid >> 3;                        // 0..31
    int tlo = tid & 7;
    int grp = oc >> 3, oc8 = oc & 7;
    int k = ksz[grp];
    const float* fw = p.fw[grp]; const float* gw = p.gw[grp];
    float fbv = p.fb[grp][oc8], gbv = p.gb[grp][oc8];
    for (int i = 0; i < 8; ++i) {
        int t = tlo + 8 * i;                  // 0..63
        size_t o = (((size_t)b * Cc + oc) * VP + n) * TP + t;
        if (n >= Nn || t >= TOUTc) { Xbf[o] = 0; continue; }
        float f = fbv, g = gbv;
        int start = t + 7 - k;                // crop [..., -58:]
        for (int ic = 0; ic < Cc; ++ic) {
            const float* fr = fw + ((size_t)oc8 * Cc + ic) * k;
            const float* gr = gw + ((size_t)oc8 * Cc + ic) * k;
            const float* xr = xs + ic * Tt + start;
            for (int dt = 0; dt < k; ++dt) {
                float xv = xr[dt];
                f = fmaf(xv, fr[dt], f);
                g = fmaf(xv, gr[dt], g);
            }
        }
        float X = tanhf(f) * (1.f / (1.f + expf(-g)));
        Xbf[o] = f2bf(X);
    }
}

// ---------------------------------------------------------------------------
// 3) WMMA mixprop hop:  Hout = 0.05*X + 0.95*(Ah @ Hin)   (per b,c slice)
//    Ah: [VP][VP] bf16, Hin/Xin/Hout: [B*C][VP][TP] bf16
//    A slab (16 x VP) staged once per block via Tensor Data Mover.
// ---------------------------------------------------------------------------
__global__ __launch_bounds__(128)
void mixprop_wmma(const unsigned short* __restrict__ Ah,
                  const unsigned short* __restrict__ Hin,
                  const unsigned short* __restrict__ Xin,
                  unsigned short* __restrict__ Hout) {
    __shared__ unsigned short ldsA[16 * VP];   // [m][k] row-major slab, 10 KB
    __shared__ unsigned short ldsB[TP * 32];   // transposed [t][k], 4 KB
    const int rowt = blockIdx.x;               // 0..19  (16-row tile of V)
    const int bc   = blockIdx.y;               // 0..1023
    const int tid  = threadIdx.x;
    const int wv   = tid >> 5;                 // wave -> 16-wide t tile
    const int lane = tid & 31;
    const int v0   = rowt * 16;
    const size_t hbase = (size_t)bc * VP * TP;

    // ---- stage the whole 16xVP Ah slab into LDS ----
#if defined(HAVE_TDM)
    if (tid < 32) {                            // wave 0 drives the TDM
        unsigned lds_off = (unsigned)(size_t)(&ldsA[0]);      // addr[31:0] = LDS offset
        unsigned long long ga = (unsigned long long)(size_t)(Ah + (size_t)v0 * VP);
        // D# group0: count=1, lds_addr, global_addr[56:0], type=2
        v4u g0;
        g0[0] = 1u;
        g0[1] = lds_off;
        g0[2] = (unsigned)ga;
        g0[3] = (unsigned)((ga >> 32) & 0x01ffffffu) | (2u << 30);
        // D# group1: data_size=2B, tensor 320x320 (stride 320), tile 320x16
        v8i g1;
        g1[0] = 1 << 16;                       // data_size = 1 (2 bytes)
        g1[1] = VP << 16;                      // tensor_dim0 low16 in [63:48]
        g1[2] = VP << 16;                      // tensor_dim1 low16 in [95:80]
        g1[3] = VP << 16;                      // tile_dim0 in [127:112]
        g1[4] = 16;                            // tile_dim1 in [143:128]
        g1[5] = VP;                            // tensor_dim0_stride low32
        g1[6] = 0;
        g1[7] = 0;
        v4i gz4 = {};
#if defined(__clang_major__) && (__clang_major__ >= 23)
        v8i gz8 = {};
        __builtin_amdgcn_tensor_load_to_lds(g0, g1, gz4, gz4, gz8, 0);
#else
        __builtin_amdgcn_tensor_load_to_lds(g0, g1, gz4, gz4, 0);
#endif
#if __has_builtin(__builtin_amdgcn_s_wait_tensorcnt)
        __builtin_amdgcn_s_wait_tensorcnt(0);
#else
        asm volatile("s_wait_tensorcnt 0x0" ::: "memory");
#endif
    }
#else
    {   // fallback: plain cooperative copy (slab is contiguous in global)
        const unsigned int* src = (const unsigned int*)(Ah + (size_t)v0 * VP);
        unsigned int* dst = (unsigned int*)ldsA;
        for (int i = tid; i < 16 * VP / 2; i += 128) dst[i] = src[i];
    }
#endif
    __syncthreads();

    v8f acc = {};
    const unsigned int* la = (const unsigned int*)ldsA;
    unsigned int* lb32 = (unsigned int*)ldsB;

    for (int kk = 0; kk < VP; kk += 32) {
        // ---- stage H K-chunk transposed: ldsB[t][k] with (k,k+1) packed ----
        {
            int kp  = tid & 15;                 // k pair -> k0 = 2*kp
            int t0  = (tid >> 4) * 8;           // 8 t per thread
            const unsigned short* s0 = Hin + hbase + (size_t)(kk + 2 * kp) * TP + t0;
            union { uint4 q; unsigned short h[8]; } r0, r1;
            r0.q = *(const uint4*)s0;
            r1.q = *(const uint4*)(s0 + TP);
            #pragma unroll
            for (int t = 0; t < 8; ++t)
                lb32[(t0 + t) * 16 + kp] =
                    (unsigned int)r0.h[t] | ((unsigned int)r1.h[t] << 16);
        }
        // prefetch next H chunk while this one is consumed
        if (kk + 32 < VP)
            __builtin_prefetch((const void*)(Hin + hbase + (size_t)(kk + 32 + (tid & 31)) * TP), 0, 3);
        __syncthreads();

        // ---- build fragments per CDNA5 16-bit WMMA VGPR layouts ----
        v8u areg, breg;
        {
            int m = lane & 15, kh = lane >> 4;
            #pragma unroll
            for (int v = 0; v < 8; ++v) {
                int grp = v >> 2, vi = v & 3;
                int kp = grp * 16 + kh * 8 + vi * 2;      // even K pair start
                areg[v] = la[m * (VP / 2) + ((kk + kp) >> 1)];
            }
            int n = lane & 15, kg = lane >> 4;
            int tcol = wv * 16 + n;
            #pragma unroll
            for (int v = 0; v < 8; ++v) {
                int kp = (v >> 2) * 16 + kg * 8 + (v & 3) * 2;
                breg[v] = lb32[tcol * 16 + (kp >> 1)];
            }
        }
        v16bf a = __builtin_bit_cast(v16bf, areg);
        v16bf b = __builtin_bit_cast(v16bf, breg);
        acc = __builtin_amdgcn_wmma_f32_16x16x32_bf16(
                  false, a, false, b, (short)0, acc, false, false);
        __syncthreads();
    }

    // ---- writeback: C/D layout: VGPR r -> M = r + 8*(lane>=16), N = lane&15
    {
        int n = lane & 15, half = lane >> 4;
        int tcol = wv * 16 + n;
        #pragma unroll
        for (int r = 0; r < 8; ++r) {
            int v = v0 + r + 8 * half;
            size_t off = hbase + (size_t)v * TP + tcol;
            float xv = bf2f(Xin[off]);
            Hout[off] = f2bf(0.05f * xv + 0.95f * acc[r]);
        }
    }
}

// ---------------------------------------------------------------------------
// 4) 1x1 mixing conv: Y[b,oc,v,t] (+)= sum_ic W[oc,coloff+ic] * H[b,ic,v,t]
//    initmode: Y = biases + x_crop + (W1+W2)[:,0:32] @ X
// ---------------------------------------------------------------------------
__global__ __launch_bounds__(64)
void conv1x1_kernel(float* __restrict__ Y,
                    const unsigned short* __restrict__ H,
                    const float* __restrict__ W1, const float* __restrict__ W2,
                    int coloff,
                    const float* __restrict__ b1, const float* __restrict__ b2,
                    const float* __restrict__ xfull, int initmode) {
    int b = blockIdx.x, v = blockIdx.y;
    __shared__ float hs[Cc * TP];     // 8 KB
    __shared__ float wsm[Cc * Cc];    // 4 KB
    int tid = threadIdx.x;
    for (int i = tid; i < Cc * TP; i += 64) {
        int ic = i >> 6, t = i & 63;
        hs[i] = bf2f(H[(((size_t)b * Cc + ic) * VP + v) * TP + t]);
    }
    for (int i = tid; i < Cc * Cc; i += 64) {
        int oc = i >> 5, ic = i & 31;
        float w = W1[oc * 96 + coloff + ic];
        if (W2) w += W2[oc * 96 + coloff + ic];
        wsm[i] = w;
    }
    __syncthreads();
    int t = tid;
    if (t >= TOUTc) return;
    for (int oc = 0; oc < Cc; ++oc) {
        float acc = 0.f;
        for (int ic = 0; ic < Cc; ++ic) acc = fmaf(wsm[oc * 32 + ic], hs[ic * TP + t], acc);
        size_t o = (((size_t)b * Cc + oc) * Nn + v) * TOUTc + t;
        if (initmode) {
            float base = b1[oc] + b2[oc] +
                         xfull[(((size_t)b * Cc + oc) * Nn + v) * Tt + (t + 6)];
            Y[o] = base + acc;
        } else {
            Y[o] += acc;
        }
    }
}

// ---------------------------------------------------------------------------
// 5) Skip conv (kernel 1x58) + residual x_skip
// ---------------------------------------------------------------------------
__global__ __launch_bounds__(64)
void skip_kernel(float* __restrict__ out,                  // [B][64][300]
                 const unsigned short* __restrict__ Xbf,
                 const float* __restrict__ xskip,
                 const float* __restrict__ sw, const float* __restrict__ sb) {
    int b = blockIdx.x, v = blockIdx.y;
    __shared__ float hs[Cc * TP];
    int tid = threadIdx.x;
    for (int i = tid; i < Cc * TP; i += 64) {
        int ic = i >> 6, t = i & 63;
        hs[i] = bf2f(Xbf[(((size_t)b * Cc + ic) * VP + v) * TP + t]);
    }
    __syncthreads();
    int oc = tid;                                          // 0..63
    float acc = sb[oc] + xskip[((size_t)b * 64 + oc) * Nn + v];
    for (int ic = 0; ic < Cc; ++ic) {
        const float* wr = sw + ((size_t)oc * Cc + ic) * TOUTc;
        const float* hr = hs + ic * TP;
        for (int t = 0; t < TOUTc; ++t) acc = fmaf(hr[t], wr[t], acc);
    }
    out[((size_t)b * 64 + oc) * Nn + v] = acc;
}

// ---------------------------------------------------------------------------
// 6) LayerNorm over (C,N,T) per batch sample
// ---------------------------------------------------------------------------
__global__ __launch_bounds__(256)
void ln_reduce(const float* __restrict__ Y, float* __restrict__ stats) {
    const size_t PER = (size_t)Cc * Nn * TOUTc;
    int b = blockIdx.x, tid = threadIdx.x;
    const float* p = Y + (size_t)b * PER;
    float s = 0.f, s2 = 0.f;
    for (size_t i = tid; i < PER; i += 256) { float v = p[i]; s += v; s2 += v * v; }
    __shared__ float sh[256], sh2[256];
    sh[tid] = s; sh2[tid] = s2;
    __syncthreads();
    for (int st = 128; st > 0; st >>= 1) {
        if (tid < st) { sh[tid] += sh[tid + st]; sh2[tid] += sh2[tid + st]; }
        __syncthreads();
    }
    if (tid == 0) {
        float mu = sh[0] / (float)PER;
        float var = sh2[0] / (float)PER - mu * mu;
        stats[b * 2 + 0] = mu;
        stats[b * 2 + 1] = var;
    }
}

__global__ __launch_bounds__(256)
void ln_apply(float* __restrict__ Y, const float* __restrict__ stats,
              const float* __restrict__ lw, const float* __restrict__ lb,
              const int* __restrict__ idx) {
    const size_t total = (size_t)Bb * Cc * Nn * TOUTc;
    size_t i = (size_t)blockIdx.x * 256 + threadIdx.x;
    if (i >= total) return;
    int t = (int)(i % TOUTc);
    size_t r = i / TOUTc;
    int v = (int)(r % Nn); r /= Nn;
    int c = (int)(r % Cc);
    int b = (int)(r / Cc);
    float mu = stats[b * 2 + 0], var = stats[b * 2 + 1];
    size_t li = ((size_t)c * Nn + idx[v]) * TOUTc + t;
    Y[i] = (Y[i] - mu) * rsqrtf(var + 1e-5f) * lw[li] + lb[li];
}

// ---------------------------------------------------------------------------
// launch
// ---------------------------------------------------------------------------
extern "C" void kernel_launch(void* const* d_in, const int* in_sizes, int n_in,
                              void* d_out, int out_size, void* d_ws, size_t ws_size,
                              hipStream_t stream) {
    (void)in_sizes; (void)n_in; (void)out_size; (void)ws_size;
    const float* x      = (const float*)d_in[0];
    const float* x_skip = (const float*)d_in[1];
    const int*   idx    = (const int*)  d_in[2];
    const float* emb1   = (const float*)d_in[3];
    const float* emb2   = (const float*)d_in[4];
    const float* lin1_w = (const float*)d_in[5];
    const float* lin1_b = (const float*)d_in[6];
    const float* lin2_w = (const float*)d_in[7];
    const float* lin2_b = (const float*)d_in[8];
    const float* skip_w = (const float*)d_in[9];
    const float* skip_b = (const float*)d_in[10];
    const float* mp1_w  = (const float*)d_in[11];
    const float* mp1_b  = (const float*)d_in[12];
    const float* mp2_w  = (const float*)d_in[13];
    const float* mp2_b  = (const float*)d_in[14];
    const float* ln_w   = (const float*)d_in[15];
    const float* ln_b   = (const float*)d_in[16];
    IncParams ip;
    // setup_inputs dict order per k: fw, fb, gw, gb  for k = 2,3,6,7
    for (int g = 0; g < 4; ++g) {
        ip.fw[g] = (const float*)d_in[17 + 4 * g + 0];
        ip.fb[g] = (const float*)d_in[17 + 4 * g + 1];
        ip.gw[g] = (const float*)d_in[17 + 4 * g + 2];
        ip.gb[g] = (const float*)d_in[17 + 4 * g + 3];
    }

    // workspace layout (256B aligned bump allocator)
    char* ws = (char*)d_ws;
    size_t off = 0;
    auto alloc = [&](size_t bytes) { char* p = ws + off; off += (bytes + 255) & ~(size_t)255; return p; };
    float*          w_n1  = (float*)alloc(Nn * DIMe * 4);
    float*          w_n2  = (float*)alloc(Nn * DIMe * 4);
    float*          w_A   = (float*)alloc((size_t)Nn * Nn * 4);
    float*          w_Am  = (float*)alloc((size_t)Nn * Nn * 4);
    float*          w_rs1 = (float*)alloc(Nn * 4);
    float*          w_rs2 = (float*)alloc(Nn * 4);
    float*          w_st  = (float*)alloc(Bb * 2 * 4);
    unsigned short* w_Ah1 = (unsigned short*)alloc((size_t)VP * VP * 2);
    unsigned short* w_Ah2 = (unsigned short*)alloc((size_t)VP * VP * 2);
    unsigned short* w_X   = (unsigned short*)alloc((size_t)Bb * Cc * VP * TP * 2);
    unsigned short* w_Ha  = (unsigned short*)alloc((size_t)Bb * Cc * VP * TP * 2);
    unsigned short* w_Hb  = (unsigned short*)alloc((size_t)Bb * Cc * VP * TP * 2);

    float* Y      = (float*)d_out;                                   // [B][32][300][58]
    float* skipO  = (float*)d_out + (size_t)Bb * Cc * Nn * TOUTc;    // [B][64][300]

    // graph constructor
    embed_kernel<<<Nn, 64, 0, stream>>>(emb1, emb2, lin1_w, lin1_b, lin2_w, lin2_b, idx, w_n1, w_n2);
    adj_kernel<<<Nn, 128, 0, stream>>>(w_n1, w_n2, w_A);
    topk_kernel<<<Nn, 128, 0, stream>>>(w_A, w_Am);
    rowsum_kernel<<<Nn, 128, 0, stream>>>(w_Am, w_rs1, w_rs2);
    writeAh_kernel<<<dim3(5, VP), 64, 0, stream>>>(w_Am, w_rs1, w_rs2, w_Ah1, w_Ah2);

    // dilated inception + gating
    inception_kernel<<<dim3(Bb, VP), 256, 0, stream>>>(x, ip, w_X);

    // Y init: biases + residual crop + hop-0 term with (W1+W2)[:,0:32]
    conv1x1_kernel<<<dim3(Bb, Nn), 64, 0, stream>>>(Y, w_X, mp1_w, mp2_w, 0, mp1_b, mp2_b, x, 1);

    dim3 mg(VP / 16, Bb * Cc);
    // graph 1 (Ah1)
    mixprop_wmma<<<mg, 128, 0, stream>>>(w_Ah1, w_X, w_X, w_Ha);
    conv1x1_kernel<<<dim3(Bb, Nn), 64, 0, stream>>>(Y, w_Ha, mp1_w, nullptr, 32, nullptr, nullptr, nullptr, 0);
    mixprop_wmma<<<mg, 128, 0, stream>>>(w_Ah1, w_Ha, w_X, w_Hb);
    conv1x1_kernel<<<dim3(Bb, Nn), 64, 0, stream>>>(Y, w_Hb, mp1_w, nullptr, 64, nullptr, nullptr, nullptr, 0);
    // graph 2 (Ah2 = rownorm(A^T + I))
    mixprop_wmma<<<mg, 128, 0, stream>>>(w_Ah2, w_X, w_X, w_Ha);
    conv1x1_kernel<<<dim3(Bb, Nn), 64, 0, stream>>>(Y, w_Ha, mp2_w, nullptr, 32, nullptr, nullptr, nullptr, 0);
    mixprop_wmma<<<mg, 128, 0, stream>>>(w_Ah2, w_Ha, w_X, w_Hb);
    conv1x1_kernel<<<dim3(Bb, Nn), 64, 0, stream>>>(Y, w_Hb, mp2_w, nullptr, 64, nullptr, nullptr, nullptr, 0);

    // skip output
    skip_kernel<<<dim3(Bb, Nn), 64, 0, stream>>>(skipO, w_X, x_skip, skip_w, skip_b);

    // layernorm
    ln_reduce<<<Bb, 256, 0, stream>>>(Y, w_st);
    const size_t total = (size_t)Bb * Cc * Nn * TOUTc;
    ln_apply<<<(unsigned)((total + 255) / 256), 256, 0, stream>>>(Y, w_st, ln_w, ln_b, idx);
}
